// LSTMNextWindowPredictor_89704686944882
// MI455X (gfx1250) — matrix operand order repaired and checked by reading, not compiled, not run on hardware
//
#include <hip/hip_runtime.h>

// ---------------------------------------------------------------------------
// LSTM (I=13,H=64,T=5) + MLP head (64->64 relu -> 13), B=262144, fp32 in/out.
// Per-wave 16-row batch tile; all matmuls via v_wmma_f32_16x16x32_bf16 with
// f32 accumulation. Weights staged once into LDS in WMMA B-fragment order.
// Bias is folded into the WMMA via a constant-1.0 column at K=13 (dead K pad)
// of the x A-fragment. Gate rows for i/f/o (and their bias) are pre-scaled by
// 0.5 so sigmoid(x) = 0.5*tanh(0.5x)+0.5 needs only v_tanh_f32 + one FMA.
// ---------------------------------------------------------------------------

typedef __attribute__((ext_vector_type(16))) __bf16 v16bf;
typedef __attribute__((ext_vector_type(8)))  float  v8f;

#define NW    4        // waves per block
#define WAVE  32
#define HH    64       // hidden size
#define TT    5        // timesteps
#define II    13       // input features
#define OO    13       // output features
#define BTOT  262144
#define HROW  72       // padded LDS row stride for h (bank-conflict break)

#if defined(__has_builtin)
#  if __has_builtin(__builtin_amdgcn_tanhf)
#    define HAS_HW_TANH 1
#  endif
#  if __has_builtin(__builtin_amdgcn_exp2f)
#    define FAST_EXP2(x) __builtin_amdgcn_exp2f(x)
#  endif
#endif
#ifndef FAST_EXP2
#  define FAST_EXP2(x) __expf((x) * 0.6931471805599453f)
#endif

union Frag {
  v16bf  v;
  uint4  q[2];
  __bf16 e[16];
};

__device__ __forceinline__ float hw_tanh(float x) {
#ifdef HAS_HW_TANH
  return __builtin_amdgcn_tanhf(x);          // v_tanh_f32 (CDNA5)
#else
  // tanh(x) = 1 - 2/(1+exp2(2*log2e*x)); stable at both extremes
  float r = __builtin_amdgcn_rcpf(1.0f + FAST_EXP2(2.885390082f * x));
  return __builtin_fmaf(-2.0f, r, 1.0f);
#endif
}
// sigmoid(2y) given y = 0.5*preactivation (gate rows pre-scaled by 0.5)
__device__ __forceinline__ float sig_from_half(float y) {
#ifdef HAS_HW_TANH
  return __builtin_fmaf(0.5f, __builtin_amdgcn_tanhf(y), 0.5f);
#else
  return __builtin_amdgcn_rcpf(1.0f + FAST_EXP2(-2.885390082f * y));
#endif
}
__device__ __forceinline__ v8f bcast8(float v) {
  v8f r;
#pragma unroll
  for (int i = 0; i < 8; ++i) r[i] = v;
  return r;
}
__device__ __forceinline__ v8f wmma_bf16(const Frag& a, const Frag& b, v8f c) {
  return __builtin_amdgcn_wmma_f32_16x16x32_bf16(false, a.v, false, b.v,
                                                 (short)0, c, false, false);
}

// B-fragment (32x16 bf16) stored half-split: slots 0..7 then slots 8..15,
// each half = 32 lanes x 8 elems (16B per lane -> minimal-phase ds_load_b128).
__device__ __forceinline__ Frag ldsFrag(const __bf16* base, int tile, int lane) {
  Frag f;
  const __bf16* p = base + tile * 512;
  f.q[0] = *(const uint4*)(p + lane * 8);
  f.q[1] = *(const uint4*)(p + 256 + lane * 8);
  return f;
}
// Compact W_ih fragment: K>=16 rows are all zero -> lanes 16..31 hold zeros.
__device__ __forceinline__ Frag ldsFragIH(const __bf16* base, int nt, int lane) {
  Frag f;
  if (lane < 16) {
    const __bf16* p = base + nt * 256;
    f.q[0] = *(const uint4*)(p + lane * 8);
    f.q[1] = *(const uint4*)(p + 128 + lane * 8);
  } else {
    f.q[0] = make_uint4(0u, 0u, 0u, 0u);
    f.q[1] = make_uint4(0u, 0u, 0u, 0u);
  }
  return f;
}

__global__ void __launch_bounds__(NW * WAVE)
lstm_head_kernel(const float* __restrict__ x,
                 const float* __restrict__ W_ih,
                 const float* __restrict__ W_hh,
                 const float* __restrict__ b_ih,
                 const float* __restrict__ b_hh,
                 const float* __restrict__ W1,
                 const float* __restrict__ b1,
                 const float* __restrict__ W2,
                 const float* __restrict__ b2,
                 float* __restrict__ out) {
  // ---- static LDS (~62 KB total) ----
  __shared__ __align__(16) __bf16 s_whh[32 * 512];   // W_hh^T frags: 32 KB
  __shared__ __align__(16) __bf16 s_wih[16 * 256];   // W_ih^T + bias row: 8 KB
  __shared__ __align__(16) __bf16 s_w1 [ 8 * 512];   // W1^T frags: 8 KB
  __shared__ __align__(16) __bf16 s_w2 [ 2 * 512];   // W2^T frags: 2 KB
  __shared__ float s_b1[64];
  __shared__ float s_b2[16];
  __shared__ __align__(16) __bf16 s_h[NW][16 * HROW]; // per-wave h (bf16)
  __shared__ __align__(16) __bf16 s_x[NW][16 * 16];   // per-wave x_t staging

  const int tid  = threadIdx.x;
  const int nthr = NW * WAVE;

  // gate scale: i/f/o rows (n<128 or n>=192) pre-scaled by 0.5; g rows by 1.0
  // ---- stage weights into LDS in WMMA B-fragment layout ----
  // element mapping (32x16 bf16 B tile): lane = ((K/8)%2)*16 + N, slot = (K/16)*8 + K%8
  for (int idx = tid; idx < 32 * 512; idx += nthr) {  // W_hh: tiles t = nt*2+kt
    int t = idx >> 9, r2 = idx & 511;
    int half = r2 >> 8, rr = r2 & 255;
    int lane = rr >> 3, sl = (half << 3) | (rr & 7);
    int kt = t & 1, nt = t >> 1;
    int k = kt * 32 + ((sl >> 3) << 4) + ((lane >> 4) << 3) + (sl & 7);
    int n = nt * 16 + (lane & 15);
    float sc = (n >= 128 && n < 192) ? 1.0f : 0.5f;
    s_whh[idx] = (__bf16)(W_hh[n * HH + k] * sc);
  }
  for (int idx = tid; idx < 16 * 256; idx += nthr) {  // W_ih + bias at k==13
    int t = idx >> 8, r2 = idx & 255;
    int half = r2 >> 7, rr = r2 & 127;
    int lane = rr >> 3, sl = (half << 3) | (rr & 7);
    int k = sl;                       // K = slot (K<16 half only)
    int n = t * 16 + lane;
    float sc = (n >= 128 && n < 192) ? 1.0f : 0.5f;
    float v = 0.0f;
    if (k < II)       v = W_ih[n * II + k] * sc;
    else if (k == II) v = (b_ih[n] + b_hh[n]) * sc;   // bias row (A has 1.0)
    s_wih[idx] = (__bf16)v;
  }
  for (int idx = tid; idx < 8 * 512; idx += nthr) {   // W1
    int t = idx >> 9, r2 = idx & 511;
    int half = r2 >> 8, rr = r2 & 255;
    int lane = rr >> 3, sl = (half << 3) | (rr & 7);
    int kt = t & 1, nt = t >> 1;
    int k = kt * 32 + ((sl >> 3) << 4) + ((lane >> 4) << 3) + (sl & 7);
    int n = nt * 16 + (lane & 15);
    s_w1[idx] = (__bf16)W1[n * HH + k];
  }
  for (int idx = tid; idx < 2 * 512; idx += nthr) {   // W2 (N padded 13->16)
    int t = idx >> 9, r2 = idx & 511;
    int half = r2 >> 8, rr = r2 & 255;
    int lane = rr >> 3, sl = (half << 3) | (rr & 7);
    int k = t * 32 + ((sl >> 3) << 4) + ((lane >> 4) << 3) + (sl & 7);
    int n = lane & 15;
    s_w2[idx] = (n < OO) ? (__bf16)W2[n * HH + k] : (__bf16)0.0f;
  }
  for (int n = tid; n < 64; n += nthr) s_b1[n] = b1[n];
  for (int n = tid; n < 16; n += nthr) s_b2[n] = (n < OO) ? b2[n] : 0.0f;
  __syncthreads();

  // ---- per-wave LSTM over a 16-row batch tile ----
  const int wid  = tid >> 5;
  const int lane = tid & 31;
  const int ln   = lane & 15;   // A-row / C-column index
  const int hi   = lane >> 4;   // lane half
  const long b0  = ((long)blockIdx.x * NW + wid) * 16;

  __bf16* hsc = s_h[wid];
  __bf16* xsc = s_x[wid];

  for (int e = lane; e < 16 * HROW; e += WAVE) hsc[e] = (__bf16)0.0f;

  const v8f vzero = bcast8(0.0f);
  v8f c[4];
#pragma unroll
  for (int j = 0; j < 4; ++j) c[j] = vzero;

  for (int t = 0; t < TT; ++t) {
    // stage x[:, t, :] tile into LDS as bf16 (16 rows x 13 cols, stride 16)
    for (int e = lane; e < 16 * II; e += WAVE) {
      int m = e / II, i = e - m * II;
      xsc[m * 16 + i] = (__bf16)x[((b0 + m) * TT + t) * II + i];
    }
    // A fragment for x_t: K<13 = data, K==13 = 1.0 (bias column), else 0
    Frag ax;
#pragma unroll
    for (int s = 0; s < 16; ++s) {
      int K = (hi << 4) + s;
      ax.e[s] = (K < II)  ? xsc[ln * 16 + K]
              : (K == II) ? (__bf16)1.0f
                          : (__bf16)0.0f;
    }
    // A fragments for h (16x64 -> two 16x32 tiles) via aligned b128 LDS loads
    Frag ah0, ah1;
    {
      const __bf16* hb = hsc + ln * HROW + (hi << 3);
      ah0.q[0] = *(const uint4*)(hb);
      ah0.q[1] = *(const uint4*)(hb + 16);
      ah1.q[0] = *(const uint4*)(hb + 32);
      ah1.q[1] = *(const uint4*)(hb + 48);
    }

#pragma unroll
    for (int j = 0; j < 4; ++j) {   // hidden-column block (cols j*16..j*16+15)
      // input contribution + bias (C starts as inline zero)
      v8f ai = wmma_bf16(ax, ldsFragIH(s_wih, 0 * 4 + j, lane), vzero);
      v8f af = wmma_bf16(ax, ldsFragIH(s_wih, 1 * 4 + j, lane), vzero);
      v8f ag = wmma_bf16(ax, ldsFragIH(s_wih, 2 * 4 + j, lane), vzero);
      v8f ao = wmma_bf16(ax, ldsFragIH(s_wih, 3 * 4 + j, lane), vzero);
      // recurrent contribution (K=64 -> 2 WMMAs each)
      ai = wmma_bf16(ah0, ldsFrag(s_whh, (0 * 4 + j) * 2 + 0, lane), ai);
      ai = wmma_bf16(ah1, ldsFrag(s_whh, (0 * 4 + j) * 2 + 1, lane), ai);
      af = wmma_bf16(ah0, ldsFrag(s_whh, (1 * 4 + j) * 2 + 0, lane), af);
      af = wmma_bf16(ah1, ldsFrag(s_whh, (1 * 4 + j) * 2 + 1, lane), af);
      ag = wmma_bf16(ah0, ldsFrag(s_whh, (2 * 4 + j) * 2 + 0, lane), ag);
      ag = wmma_bf16(ah1, ldsFrag(s_whh, (2 * 4 + j) * 2 + 1, lane), ag);
      ao = wmma_bf16(ah0, ldsFrag(s_whh, (3 * 4 + j) * 2 + 0, lane), ao);
      ao = wmma_bf16(ah1, ldsFrag(s_whh, (3 * 4 + j) * 2 + 1, lane), ao);

      // gate nonlinearities + state update; write h back in C-layout
      // ai/af/ao hold 0.5*preactivation (pre-scaled weights+bias)
#pragma unroll
      for (int r = 0; r < 8; ++r) {
        float iv = sig_from_half(ai[r]);
        float fv = sig_from_half(af[r]);
        float ov = sig_from_half(ao[r]);
        float gv = hw_tanh(ag[r]);
        float cv = __builtin_fmaf(fv, c[j][r], iv * gv);
        c[j][r] = cv;
        float hv = ov * hw_tanh(cv);
        hsc[(r + (hi << 3)) * HROW + j * 16 + ln] = (__bf16)hv;
      }
    }
  }

  // ---- head: z = relu(h @ W1^T + b1); out = z @ W2^T + b2 ----
  Frag fh0, fh1;
  {
    const __bf16* hb = hsc + ln * HROW + (hi << 3);
    fh0.q[0] = *(const uint4*)(hb);
    fh0.q[1] = *(const uint4*)(hb + 16);
    fh1.q[0] = *(const uint4*)(hb + 32);
    fh1.q[1] = *(const uint4*)(hb + 48);
  }
#pragma unroll
  for (int nt = 0; nt < 4; ++nt) {
    v8f az = bcast8(s_b1[nt * 16 + ln]);
    az = wmma_bf16(fh0, ldsFrag(s_w1, nt * 2 + 0, lane), az);
    az = wmma_bf16(fh1, ldsFrag(s_w1, nt * 2 + 1, lane), az);
#pragma unroll
    for (int r = 0; r < 8; ++r) {
      float zv = fmaxf(az[r], 0.0f);
      hsc[(r + (hi << 3)) * HROW + nt * 16 + ln] = (__bf16)zv;  // z overwrites h
    }
  }
  Frag fz0, fz1;
  {
    const __bf16* hb = hsc + ln * HROW + (hi << 3);
    fz0.q[0] = *(const uint4*)(hb);
    fz0.q[1] = *(const uint4*)(hb + 16);
    fz1.q[0] = *(const uint4*)(hb + 32);
    fz1.q[1] = *(const uint4*)(hb + 48);
  }
  v8f aout = bcast8(s_b2[ln]);
  aout = wmma_bf16(fz0, ldsFrag(s_w2, 0, lane), aout);
  aout = wmma_bf16(fz1, ldsFrag(s_w2, 1, lane), aout);

  if (ln < OO) {
#pragma unroll
    for (int r = 0; r < 8; ++r) {
      out[(b0 + r + (hi << 3)) * OO + ln] = aout[r];
    }
  }
}

extern "C" void kernel_launch(void* const* d_in, const int* in_sizes, int n_in,
                              void* d_out, int out_size, void* d_ws, size_t ws_size,
                              hipStream_t stream) {
  const float* x    = (const float*)d_in[0];
  const float* W_ih = (const float*)d_in[1];
  const float* W_hh = (const float*)d_in[2];
  const float* b_ih = (const float*)d_in[3];
  const float* b_hh = (const float*)d_in[4];
  const float* W1   = (const float*)d_in[5];
  const float* b1   = (const float*)d_in[6];
  const float* W2   = (const float*)d_in[7];
  const float* b2   = (const float*)d_in[8];
  float* out = (float*)d_out;

  dim3 grid(BTOT / (16 * NW));   // 4096 blocks, each covers 64 batch rows
  dim3 block(NW * WAVE);         // 128 threads = 4 waves
  hipLaunchKernelGGL(lstm_head_kernel, grid, block, 0, stream,
                     x, W_ih, W_hh, b_ih, b_hh, W1, b1, W2, b2, out);
}
